// GPTSelfAttention1D_3942779977861
// MI455X (gfx1250) — matrix operand order repaired and checked by reading, not compiled, not run on hardware
//
#include <hip/hip_runtime.h>

#define S_LEN   2048
#define DMODEL  2048
#define NQKV    6144
#define HDIM    128
#define NHEADS  16
#define BATCH   4

typedef __attribute__((ext_vector_type(16))) __bf16 v16bf;
typedef __attribute__((ext_vector_type(8)))  float  v8f;
typedef int v4i_t __attribute__((vector_size(16)));

union FragAB { v16bf v; uint4 q[2]; };

__device__ __forceinline__ unsigned short f2bf(float f) {
  unsigned int u = __float_as_uint(f);
  u += 0x7fffu + ((u >> 16) & 1u);          // round-to-nearest-even
  return (unsigned short)(u >> 16);
}

__device__ __forceinline__ v8f wmma_bf16(const FragAB& a, const FragAB& b, v8f c) {
  return __builtin_amdgcn_wmma_f32_16x16x32_bf16(false, a.v, false, b.v,
                                                 (short)0, c, false, false);
}

// ---------------------------------------------------------------------------
// CDNA5 async global->LDS copy (ASYNCcnt), with portable fallback.
// Builtin signature (from hipcc diagnostic): (v4i AS1*, v4i AS3*, Imm, Imm).
// ---------------------------------------------------------------------------
__device__ __forceinline__ void async_copy16(const ushort* gsrc, ushort* ldst) {
#if __has_builtin(__builtin_amdgcn_global_load_async_to_lds_b128)
  __builtin_amdgcn_global_load_async_to_lds_b128(
      (__attribute__((address_space(1))) v4i_t*)gsrc,
      (__attribute__((address_space(3))) v4i_t*)ldst, 0, 0);
#else
  *(uint4*)ldst = *(const uint4*)gsrc;
#endif
}

__device__ __forceinline__ void wait_async0() {
#if __has_builtin(__builtin_amdgcn_s_wait_asynccnt)
  __builtin_amdgcn_s_wait_asynccnt(0);
#else
  asm volatile("s_wait_asynccnt 0" ::: "memory");
#endif
}

// Stage a 128-row x 32-col bf16 tile (row-major, 16B chunks) into padded LDS
// [128][40]. 512 chunks / 256 threads = 2 async ops per thread.
__device__ __forceinline__ void stage128x32(const ushort* __restrict__ g,
                                            long long stride, ushort* l,
                                            int tid, int kt) {
  const int r0 = tid >> 2;
  const int c0 = (tid & 3) * 8;
  async_copy16(g + (long long)r0 * stride + kt + c0, l + r0 * 40 + c0);
  async_copy16(g + (long long)(r0 + 64) * stride + kt + c0, l + (r0 + 64) * 40 + c0);
}

// ---------------------------------------------------------------------------
// fp32 -> bf16 bulk conversion (vectorized x4)
// ---------------------------------------------------------------------------
__global__ void cvt_f32_bf16_x4(const float4* __restrict__ in,
                                ushort4* __restrict__ out, int n4) {
  int i = blockIdx.x * blockDim.x + threadIdx.x;
  int stride = gridDim.x * blockDim.x;
  for (; i < n4; i += stride) {
    float4 f = in[i];
    ushort4 o;
    o.x = f2bf(f.x); o.y = f2bf(f.y); o.z = f2bf(f.z); o.w = f2bf(f.w);
    out[i] = o;
  }
}

// ---------------------------------------------------------------------------
// fp32 [R,C] -> bf16 [C,R] transpose-convert. Block (32,8), 32x32 LDS tile.
// Coalesced reads and writes; weights become [N,K] so GEMM B-tiles are
// row-major and can be staged with async-to-LDS (no per-element transpose).
// ---------------------------------------------------------------------------
__global__ __launch_bounds__(256)
void cvt_transpose_bf16(const float* __restrict__ in, ushort* __restrict__ out,
                        int R, int C) {
  __shared__ ushort t[32][33];
  const int c0 = blockIdx.x * 32;
  const int r0 = blockIdx.y * 32;
  const int tx = threadIdx.x, ty = threadIdx.y;
  for (int i = 0; i < 4; i++) {
    int r = r0 + ty + i * 8;
    t[ty + i * 8][tx] = f2bf(in[(long long)r * C + c0 + tx]);
  }
  __syncthreads();
  for (int i = 0; i < 4; i++) {
    int cc = c0 + ty + i * 8;
    out[(long long)cc * R + r0 + tx] = t[tx][ty + i * 8];
  }
}

// ---------------------------------------------------------------------------
// QKV projection: x[8192,2048]bf16 @ w_qkv^T[6144,2048]bf16 + bias, scatter to
// Q,K [B,H,S,hd] and V transposed [B,H,hd,S], all bf16.
// 128x128 C-tile, 8 waves x (32x64). Double-buffered async LDS staging.
// ---------------------------------------------------------------------------
__global__ __launch_bounds__(256)
void qkv_gemm(const ushort* __restrict__ xb, const ushort* __restrict__ wt,
              const float* __restrict__ bias,
              ushort* __restrict__ Q, ushort* __restrict__ K,
              ushort* __restrict__ Vt) {
  __shared__ alignas(16) ushort lA[2][128 * 40];
  __shared__ alignas(16) ushort lB[2][128 * 40];

  const int tid  = threadIdx.x;
  const int lane = tid & 31;
  const int wave = tid >> 5;
  const int g  = lane >> 4;
  const int ln = lane & 15;

  const int m0 = blockIdx.y * 128;
  const int n0 = blockIdx.x * 128;
  const int wm = (wave >> 1) * 32;
  const int wn = (wave & 1) * 64;

  const ushort* Ag = xb + (long long)m0 * DMODEL;
  const ushort* Bg = wt + (long long)n0 * DMODEL;

  v8f acc[2][4];
  for (int a = 0; a < 2; a++)
    for (int b = 0; b < 4; b++)
      for (int e = 0; e < 8; e++) acc[a][b][e] = 0.0f;

  // prologue: stage tile 0 into buffer 0
  stage128x32(Ag, DMODEL, lA[0], tid, 0);
  stage128x32(Bg, DMODEL, lB[0], tid, 0);

  for (int kt = 0; kt < DMODEL; kt += 32) {
    const int cur = (kt >> 5) & 1;
    wait_async0();          // my wave's tile-kt copies done
    __syncthreads();        // everyone's done; prev compute finished
    if (kt + 32 < DMODEL) { // prefetch next tile into other buffer
      stage128x32(Ag, DMODEL, lA[cur ^ 1], tid, kt + 32);
      stage128x32(Bg, DMODEL, lB[cur ^ 1], tid, kt + 32);
    }

    FragAB afr[2], bfr[4];
    for (int a = 0; a < 2; a++) {
      const ushort* p = &lA[cur][(wm + a * 16 + ln) * 40 + g * 8];
      afr[a].q[0] = *(const uint4*)p;
      afr[a].q[1] = *(const uint4*)(p + 16);
    }
    for (int b = 0; b < 4; b++) {
      const ushort* p = &lB[cur][(wn + b * 16 + ln) * 40 + g * 8];
      bfr[b].q[0] = *(const uint4*)p;
      bfr[b].q[1] = *(const uint4*)(p + 16);
    }
    for (int a = 0; a < 2; a++)
      for (int b = 0; b < 4; b++)
        acc[a][b] = wmma_bf16(afr[a], bfr[b], acc[a][b]);
  }

  // epilogue: bias + scatter into Q / K / V^T
  for (int a = 0; a < 2; a++)
    for (int b = 0; b < 4; b++)
      for (int r = 0; r < 8; r++) {
        int gm = m0 + wm + a * 16 + r + g * 8;       // flat b*S + s
        int gn = n0 + wn + b * 16 + ln;              // 0..6143
        float val = acc[a][b][r] + bias[gn];
        unsigned short bv = f2bf(val);
        int bb = gm >> 11, s = gm & 2047;
        int h = gn / 384, j = gn % 384;
        long long head = (long long)(bb * NHEADS + h);
        if (j < HDIM)
          Q[(head * S_LEN + s) * HDIM + j] = bv;
        else if (j < 2 * HDIM)
          K[(head * S_LEN + s) * HDIM + (j - HDIM)] = bv;
        else
          Vt[(head * HDIM + (j - 2 * HDIM)) * S_LEN + s] = bv;
      }
}

// ---------------------------------------------------------------------------
// Flash attention, causal. Grid (S/128, H, B), 256 threads.
// Each wave owns 16 query rows; streams 32-key chunks.
// ---------------------------------------------------------------------------
__global__ __launch_bounds__(256)
void attn_kernel(const ushort* __restrict__ Q, const ushort* __restrict__ K,
                 const ushort* __restrict__ Vt, ushort* __restrict__ ctxb) {
  __shared__ alignas(16) ushort pbuf[8][16][32];    // per-wave P staging

  const int tid  = threadIdx.x;
  const int lane = tid & 31;
  const int wave = tid >> 5;
  const int g  = lane >> 4;
  const int ln = lane & 15;

  const int h = blockIdx.y;
  const int b = blockIdx.z;
  const int q0 = blockIdx.x * 128 + wave * 16;

  const long long qkbase = (long long)(b * NHEADS + h) * S_LEN * HDIM;
  const long long vtbase = (long long)(b * NHEADS + h) * HDIM * S_LEN;

  // preload Q A-fragments (4 k-steps over hd=128)
  FragAB qf[4];
  {
    const ushort* qrow = Q + qkbase + (long long)(q0 + ln) * HDIM;
    for (int s4 = 0; s4 < 4; s4++) {
      const ushort* p = qrow + s4 * 32 + g * 8;
      qf[s4].q[0] = *(const uint4*)p;
      qf[s4].q[1] = *(const uint4*)(p + 16);
    }
  }

  v8f acc[8];
  float mrow[8], srow[8], corr[8];
  for (int j = 0; j < 8; j++)
    for (int e = 0; e < 8; e++) acc[j][e] = 0.0f;
  for (int r = 0; r < 8; r++) { mrow[r] = -3.0e38f; srow[r] = 0.0f; }

  const float scale = 0.08838834764831845f;   // 1/sqrt(128)
  const int nchunks = (q0 + 15) / 32 + 1;     // causal bound for this wave

  for (int ch = 0; ch < nchunks; ch++) {
    const int kv0 = ch * 32;

    // ---- scores: two 16(q)x16(kv) tiles, contracted over hd=128 ----
    v8f s0, s1;
    for (int e = 0; e < 8; e++) { s0[e] = 0.0f; s1[e] = 0.0f; }
    for (int s4 = 0; s4 < 4; s4++) {
      FragAB kf0, kf1;
      const ushort* k0p = K + qkbase + (long long)(kv0 + ln) * HDIM + s4 * 32 + g * 8;
      kf0.q[0] = *(const uint4*)k0p;
      kf0.q[1] = *(const uint4*)(k0p + 16);
      const ushort* k1p = k0p + 16 * HDIM;
      kf1.q[0] = *(const uint4*)k1p;
      kf1.q[1] = *(const uint4*)(k1p + 16);
      s0 = wmma_bf16(qf[s4], kf0, s0);
      s1 = wmma_bf16(qf[s4], kf1, s1);
    }

    // ---- online softmax (rows live across 16-lane groups) ----
    for (int r = 0; r < 8; r++) {
      int qi  = q0 + r + g * 8;
      int kj0 = kv0 + ln;
      int kj1 = kv0 + 16 + ln;
      float v0 = (kj0 <= qi) ? s0[r] * scale : -10000.0f;
      float v1 = (kj1 <= qi) ? s1[r] * scale : -10000.0f;
      float mx = fmaxf(v0, v1);
      for (int msk = 1; msk < 16; msk <<= 1)
        mx = fmaxf(mx, __shfl_xor(mx, msk, 32));
      float nm = fmaxf(mrow[r], mx);
      float cc = __expf(mrow[r] - nm);
      float p0 = __expf(v0 - nm);
      float p1 = __expf(v1 - nm);
      float ls = p0 + p1;
      for (int msk = 1; msk < 16; msk <<= 1)
        ls += __shfl_xor(ls, msk, 32);
      srow[r] = srow[r] * cc + ls;
      mrow[r] = nm;
      corr[r] = cc;
      int m = r + g * 8;
      pbuf[wave][m][ln]      = f2bf(p0);
      pbuf[wave][m][16 + ln] = f2bf(p1);
    }

    // wave-local LDS RAW: same wave wrote pbuf, LDS ops are in-order per wave
    asm volatile("s_wait_dscnt 0" ::: "memory");

    FragAB pf;
    {
      const ushort* pp = &pbuf[wave][ln][g * 8];
      pf.q[0] = *(const uint4*)pp;
      pf.q[1] = *(const uint4*)(pp + 16);
    }

    // ---- P x V: 8 hd-column tiles, K-dim = this 32-key chunk ----
    for (int j = 0; j < 8; j++) {
      for (int r = 0; r < 8; r++) acc[j][r] *= corr[r];
      FragAB vf;
      const ushort* vp = Vt + vtbase + (long long)(16 * j + ln) * S_LEN + kv0 + g * 8;
      vf.q[0] = *(const uint4*)vp;
      vf.q[1] = *(const uint4*)(vp + 16);
      acc[j] = wmma_bf16(pf, vf, acc[j]);
    }
  }

  // ---- normalize + write ctx (bf16, [B,S,D] layout for out-proj GEMM) ----
  for (int j = 0; j < 8; j++)
    for (int r = 0; r < 8; r++) {
      int m = r + g * 8;
      float o = acc[j][r] / srow[r];
      ctxb[(long long)(b * S_LEN + q0 + m) * DMODEL + h * HDIM + 16 * j + ln] = f2bf(o);
    }
}

// ---------------------------------------------------------------------------
// Output projection: ctx[8192,2048]bf16 @ w_out^T[2048,2048]bf16 + b_out -> f32
// ---------------------------------------------------------------------------
__global__ __launch_bounds__(256)
void out_gemm(const ushort* __restrict__ cb, const ushort* __restrict__ wt,
              const float* __restrict__ bias, float* __restrict__ out) {
  __shared__ alignas(16) ushort lA[2][128 * 40];
  __shared__ alignas(16) ushort lB[2][128 * 40];

  const int tid  = threadIdx.x;
  const int lane = tid & 31;
  const int wave = tid >> 5;
  const int g  = lane >> 4;
  const int ln = lane & 15;

  const int m0 = blockIdx.y * 128;
  const int n0 = blockIdx.x * 128;
  const int wm = (wave >> 1) * 32;
  const int wn = (wave & 1) * 64;

  const ushort* Ag = cb + (long long)m0 * DMODEL;
  const ushort* Bg = wt + (long long)n0 * DMODEL;

  v8f acc[2][4];
  for (int a = 0; a < 2; a++)
    for (int b = 0; b < 4; b++)
      for (int e = 0; e < 8; e++) acc[a][b][e] = 0.0f;

  stage128x32(Ag, DMODEL, lA[0], tid, 0);
  stage128x32(Bg, DMODEL, lB[0], tid, 0);

  for (int kt = 0; kt < DMODEL; kt += 32) {
    const int cur = (kt >> 5) & 1;
    wait_async0();
    __syncthreads();
    if (kt + 32 < DMODEL) {
      stage128x32(Ag, DMODEL, lA[cur ^ 1], tid, kt + 32);
      stage128x32(Bg, DMODEL, lB[cur ^ 1], tid, kt + 32);
    }

    FragAB afr[2], bfr[4];
    for (int a = 0; a < 2; a++) {
      const ushort* p = &lA[cur][(wm + a * 16 + ln) * 40 + g * 8];
      afr[a].q[0] = *(const uint4*)p;
      afr[a].q[1] = *(const uint4*)(p + 16);
    }
    for (int b = 0; b < 4; b++) {
      const ushort* p = &lB[cur][(wn + b * 16 + ln) * 40 + g * 8];
      bfr[b].q[0] = *(const uint4*)p;
      bfr[b].q[1] = *(const uint4*)(p + 16);
    }
    for (int a = 0; a < 2; a++)
      for (int b = 0; b < 4; b++)
        acc[a][b] = wmma_bf16(afr[a], bfr[b], acc[a][b]);
  }

  for (int a = 0; a < 2; a++)
    for (int b = 0; b < 4; b++)
      for (int r = 0; r < 8; r++) {
        int gm = m0 + wm + a * 16 + r + g * 8;
        int gn = n0 + wn + b * 16 + ln;
        out[(long long)gm * DMODEL + gn] = acc[a][b][r] + bias[gn];
      }
}

// ---------------------------------------------------------------------------
// Host launcher
// ---------------------------------------------------------------------------
extern "C" void kernel_launch(void* const* d_in, const int* in_sizes, int n_in,
                              void* d_out, int out_size, void* d_ws, size_t ws_size,
                              hipStream_t stream) {
  (void)in_sizes; (void)n_in; (void)out_size; (void)ws_size;

  const float* x     = (const float*)d_in[0];
  const float* w_qkv = (const float*)d_in[1];
  const float* b_qkv = (const float*)d_in[2];
  const float* w_out = (const float*)d_in[3];
  const float* b_out = (const float*)d_in[4];
  float* out = (float*)d_out;

  char* ws = (char*)d_ws;
  ushort* xb    = (ushort*)(ws + 0);            //  32 MB  [8192,2048]
  ushort* wqkvb = (ushort*)(ws + 33554432LL);   //  24 MB  [6144,2048] (W^T)
  ushort* woutb = (ushort*)(ws + 58720256LL);   //   8 MB  [2048,2048] (W^T)
  ushort* Qb    = (ushort*)(ws + 67108864LL);   //  32 MB  [B,H,S,hd]
  ushort* Kb    = (ushort*)(ws + 100663296LL);  //  32 MB  [B,H,S,hd]
  ushort* Vtb   = (ushort*)(ws + 134217728LL);  //  32 MB  [B,H,hd,S]
  ushort* ctxb  = (ushort*)(ws + 167772160LL);  //  32 MB  [8192,2048]

  // x: flat fp32 -> bf16
  {
    long long n = (long long)BATCH * S_LEN * DMODEL;
    int n4 = (int)(n / 4);
    int blocks = (n4 + 255) / 256; if (blocks > 4096) blocks = 4096;
    cvt_f32_bf16_x4<<<blocks, 256, 0, stream>>>((const float4*)x, (ushort4*)xb, n4);
  }
  // weights: fp32 [K,N] -> bf16 [N,K]
  cvt_transpose_bf16<<<dim3(NQKV / 32, DMODEL / 32), dim3(32, 8), 0, stream>>>(
      w_qkv, wqkvb, DMODEL, NQKV);
  cvt_transpose_bf16<<<dim3(DMODEL / 32, DMODEL / 32), dim3(32, 8), 0, stream>>>(
      w_out, woutb, DMODEL, DMODEL);

  qkv_gemm<<<dim3(NQKV / 128, (BATCH * S_LEN) / 128), 256, 0, stream>>>(
      xb, wqkvb, b_qkv, Qb, Kb, Vtb);

  attn_kernel<<<dim3(S_LEN / 128, NHEADS, BATCH), 256, 0, stream>>>(
      Qb, Kb, Vtb, ctxb);

  out_gemm<<<dim3(DMODEL / 128, (BATCH * S_LEN) / 128), 256, 0, stream>>>(
      ctxb, woutb, b_out, out);
}
